// FFMLayer_57535381897662
// MI455X (gfx1250) — matrix operand order repaired and checked by reading, not compiled, not run on hardware
//
#include <hip/hip_runtime.h>

typedef __attribute__((ext_vector_type(16))) _Float16 v16h;
typedef __attribute__((ext_vector_type(8)))  float    v8f;

constexpr int kF    = 20;
constexpr int kV    = 10000;
constexpr int kVTOT = kF * kV;     // 200000
constexpr int kD    = 16;
constexpr int kDD   = 13;

constexpr int kNumPairs = kF * (kF - 1) / 2;   // 190
constexpr int kNumDuos  = kNumPairs / 2;       // 95 (two pairs per WMMA)

constexpr int kWavesPerGroup = 4;              // 95 duos split across 4 waves
constexpr int kBlockThreads  = 32 * kWavesPerGroup;

// ---- compile-time duo table: two (i,j) pairs per WMMA --------------------
struct Duo { unsigned char i1, j1, i2, j2; };
struct DuoTable { Duo d[kNumDuos]; };

constexpr DuoTable makeDuoTable() {
  DuoTable a{};
  int p = 0;
  Duo tmp{};
  bool have = false;
  for (int i = 0; i < kF - 1; ++i)
    for (int j = i + 1; j < kF; ++j) {
      if (!have) {
        tmp.i1 = (unsigned char)i; tmp.j1 = (unsigned char)j; have = true;
      } else {
        tmp.i2 = (unsigned char)i; tmp.j2 = (unsigned char)j;
        a.d[p++] = tmp; have = false;
      }
    }
  return a;
}
__constant__ DuoTable kDuoTab = makeDuoTable();

__device__ __forceinline__ void cvt8(v16h& dst, int base, float4 a, float4 b) {
  dst[base + 0] = (_Float16)a.x; dst[base + 1] = (_Float16)a.y;
  dst[base + 2] = (_Float16)a.z; dst[base + 3] = (_Float16)a.w;
  dst[base + 4] = (_Float16)b.x; dst[base + 5] = (_Float16)b.y;
  dst[base + 6] = (_Float16)b.z; dst[base + 7] = (_Float16)b.w;
}

// One block = 16 samples. 4 waves each process ~24 of the 95 pair-duos with
// V_WMMA_F32_16X16X32_F16 (pair1 -> K=0..15, pair2 -> K=16..31). The diagonal
// of the f32 accumulator is the per-sample cross sum. Unrolled x2 with two
// accumulators: 16 global_load_b128 in flight per wave, no WMMA RAW chain.
__global__ __launch_bounds__(kBlockThreads) void ffm_fused(
    const float* __restrict__ dense, const int* __restrict__ sparse,
    const float* __restrict__ tables, const float* __restrict__ w_dense,
    const float* __restrict__ w_sparse, const float* __restrict__ bias,
    float* __restrict__ out)
{
  __shared__ int   s_idx[16 * kF];                // global vocab index per (sample, field)
  __shared__ float s_part[kWavesPerGroup * 16];   // partial cross per wave per sample

  const int tid  = threadIdx.x;
  const int lane = tid & 31;
  const int q    = tid >> 5;        // wave within block (0..3)
  const int s    = lane & 15;       // my sample within the 16-sample group
  const int h    = lane >> 4;       // lane half (selects K-halves / pair for B)
  const int b0   = blockIdx.x * 16; // first sample of this group

  // Stage indices (field offsets folded in) to LDS.
  for (int t = tid; t < 16 * kF; t += kBlockThreads) {
    const int samp = t / kF, f = t - samp * kF;
    s_idx[t] = sparse[(size_t)(b0 + samp) * kF + f] + f * kV;
  }
  __syncthreads();

  const int* my_idx = &s_idx[s * kF];

  // u_p[s] = tables[i_p, idx[s, j_p]] (A rows), v_p[s] = tables[j_p, idx[s, i_p]] (B cols)
  auto duoBody = [&](int t, v8f& c) {
    const Duo dd = kDuoTab.d[t];
    const int i1 = dd.i1, j1 = dd.j1, i2 = dd.i2, j2 = dd.j2;

    const int gj1 = my_idx[j1];
    const int gj2 = my_idx[j2];
    const int gip = my_idx[h ? i2 : i1];

    const float4* pu1 = (const float4*)(tables + ((size_t)(h ? i2 : i1) * kVTOT + (h ? gj2 : gj1)) * kD);
    const float4* pv  = (const float4*)(tables + ((size_t)(h ? j2 : j1) * kVTOT + gip) * kD);

    const float4 u1a = pu1[0], u1b = pu1[1];   // full u row of my half's pair
    const float4 u2a = pu1[2], u2b = pu1[3];
    const float4 v0  = pv[0],  v1  = pv[1];    // full v row of my half's pair
    const float4 v2  = pv[2],  v3  = pv[3];

    // A (16x32 f16): lanes<16 hold K=0..15 (pair1 u), lanes>=16 hold K=16..31 (pair2 u).
    v16h A, Bm;
    cvt8(A, 0, u1a, u1b);
    cvt8(A, 8, u2a, u2b);
    // B (32x16 f16): lanes<16 hold K=0..15 (pair1 v), lanes>=16 hold K=16..31 (pair2 v).
    cvt8(Bm, 0, v0, v1);
    cvt8(Bm, 8, v2, v3);

    c = __builtin_amdgcn_wmma_f32_16x16x32_f16(
            /*neg_a=*/false, A, /*neg_b=*/false, Bm,
            /*c_mod=*/(short)0, c, /*reuse_a=*/false, /*reuse_b=*/false);
  };

  v8f c0 = {}, c1 = {};
  // Wave q owns duos {q, q+4, q+8, ...}; process two per iteration.
  int t = q;
  for (; t + kWavesPerGroup < kNumDuos; t += 2 * kWavesPerGroup) {
    duoBody(t, c0);
    duoBody(t + kWavesPerGroup, c1);
  }
  if (t < kNumDuos) duoBody(t, c0);

  // Diagonal D[s][s]: sample s<8 -> lane s, c[s]; sample s>=8 -> lane 16+s, c[s-8].
  const bool own = (lane < 8) || (lane >= 24);
  const int  r   = lane & 7;
  float diag = 0.f;
#pragma unroll
  for (int k = 0; k < 8; ++k) if (r == k) diag = c0[k] + c1[k];
  if (own) s_part[q * 16 + ((lane < 8) ? lane : (lane - 16))] = diag;
  __syncthreads();

  // Finalize: linear term + cross, sigmoid, store.
  if (tid < 16) {
    float cross = 0.f;
#pragma unroll
    for (int w = 0; w < kWavesPerGroup; ++w) cross += s_part[w * 16 + tid];

    float lin = bias[0];
    const float* dr = dense + (size_t)(b0 + tid) * kDD;
#pragma unroll
    for (int k = 0; k < kDD; ++k) lin += dr[k] * w_dense[k];
#pragma unroll
    for (int f = 0; f < kF; ++f) lin += w_sparse[s_idx[tid * kF + f]];

    const float x = lin + cross;
    out[b0 + tid] = 1.0f / (1.0f + __expf(-x));
  }
}

extern "C" void kernel_launch(void* const* d_in, const int* in_sizes, int n_in,
                              void* d_out, int out_size, void* d_ws, size_t ws_size,
                              hipStream_t stream) {
  const float* dense  = (const float*)d_in[0];  // (B, 13)
  const int*   sparse = (const int*)  d_in[1];  // (B, 20)
  const float* tables = (const float*)d_in[2];  // (20, 200000, 16)
  const float* wd     = (const float*)d_in[3];  // (13, 1)
  const float* ws     = (const float*)d_in[4];  // (200000, 1)
  const float* bias   = (const float*)d_in[5];  // (1,)
  float* out = (float*)d_out;                   // (B, 1) f32

  const int B      = in_sizes[1] / kF;          // 16384
  const int groups = B / 16;                    // 1024 blocks, 16 samples each

  ffm_fused<<<groups, kBlockThreads, 0, stream>>>(dense, sparse, tables, wd, ws, bias, out);
}
